// Mamba2_58179626992138
// MI455X (gfx1250) — matrix operand order repaired and checked by reading, not compiled, not run on hardware
//
#include <hip/hip_runtime.h>

typedef _Float16 half_t;
typedef __attribute__((ext_vector_type(16))) _Float16 v16h;
typedef __attribute__((ext_vector_type(8)))  float    v8f;

// Problem constants (from reference)
#define BATCH 4
#define CDIM  96
#define DINN  96
#define NST   8
#define KDIR  4
#define RNK   6
#define HDIM  64
#define WDIM  64
#define LDIM  4096          // HDIM*WDIM
#define CDBL  22            // RNK + 2*NST
#define NCHUNK 64
#define CHLEN  64           // LDIM / NCHUNK
#define KDIMC 96            // reduction depth of all three GEMMs

// ---------------------------------------------------------------------------
// Layout transforms
// ---------------------------------------------------------------------------
__global__ void nchw_to_nlc(const float* __restrict__ in, float* __restrict__ out) {
    size_t idx = (size_t)blockIdx.x * blockDim.x + threadIdx.x;  // over B*L*C
    if (idx >= (size_t)BATCH * LDIM * CDIM) return;
    int c = idx % CDIM;
    int l = (idx / CDIM) % LDIM;
    int b = idx / ((size_t)CDIM * LDIM);
    out[idx] = in[((size_t)b * CDIM + c) * LDIM + l];
}

__global__ void nlc_to_nchw(const float* __restrict__ in, float* __restrict__ out) {
    size_t idx = (size_t)blockIdx.x * blockDim.x + threadIdx.x;  // over B*C*L
    if (idx >= (size_t)BATCH * LDIM * CDIM) return;
    int l = idx % LDIM;
    int c = (idx / LDIM) % CDIM;
    int b = idx / ((size_t)CDIM * LDIM);
    out[idx] = in[((size_t)b * LDIM + l) * CDIM + c];
}

// ---------------------------------------------------------------------------
// Weight prep: transpose, split f32->(hi,lo) f16, and PRE-SWIZZLE into the
// per-lane WMMA B-fragment layout so the GEMM loads one contiguous 32-byte v16h per lane per fragment.
// Logical B[i][j] = W[j][i]  (i = reduction 0..c-1, j = out col, zero pad j>=r)
// Packed index: frag = (i/32)*(npad/16) + (j/16); lane = ((i>>4)&1)*16 + (j&15);
//               elem = i & 15;  dst = frag*512 + lane*16 + elem
// ---------------------------------------------------------------------------
__device__ __forceinline__ int bpack_idx(int i, int j, int npad) {
    int frag = (i >> 5) * (npad >> 4) + (j >> 4);
    int lane = ((i >> 4) & 1) * 16 + (j & 15);
    int elem = i & 15;
    return frag * 512 + lane * 16 + elem;
}

__global__ void prep_transpose_split(const float* __restrict__ W,
                                     half_t* __restrict__ hi, half_t* __restrict__ lo,
                                     int r, int c, int npad) {
    int idx = blockIdx.x * blockDim.x + threadIdx.x;
    if (idx >= c * npad) return;
    int i = idx / npad, j = idx % npad;
    float v = (j < r) ? W[(size_t)j * c + i] : 0.f;
    half_t h = (half_t)v;
    int dst = bpack_idx(i, j, npad);
    hi[dst] = h;
    lo[dst] = (half_t)(v - (float)h);
}

// xproj_w (K, 22, 96) -> per-k packed B (96 x 32 padded)
__global__ void prep_w2(const float* __restrict__ W,
                        half_t* __restrict__ hi, half_t* __restrict__ lo) {
    int idx = blockIdx.x * blockDim.x + threadIdx.x;  // K*96*32
    if (idx >= KDIR * DINN * 32) return;
    int j = idx % 32;
    int i = (idx / 32) % DINN;
    int k = idx / (32 * DINN);
    float v = (j < CDBL) ? W[((size_t)k * CDBL + j) * DINN + i] : 0.f;
    half_t h = (half_t)v;
    int dst = k * (DINN * 32) + bpack_idx(i, j, 32);
    hi[dst] = h;
    lo[dst] = (half_t)(v - (float)h);
}

// ---------------------------------------------------------------------------
// Split-f16 WMMA GEMM:  C[M x Nvalid] = A[M x 96](f32) * B[96 x npad](split f16)
// Each wave computes a 16 x (NT*16) strip: the A fragment (and its f32->hi/lo
// split) is loaded/converted ONCE per k-step and reused across NT tile cols.
// WMMAs are issued in t-sweeps so consecutive WMMAs hit DIFFERENT accumulators
// (dependent pairs are NT apart -> the 5-slot WMMA RAW hazard window is filled
// with useful matrix work instead of v_nops).
// ---------------------------------------------------------------------------
template <int NT>
__global__ __launch_bounds__(256) void gemm_splitf16(
    const float* __restrict__ A, int lda,
    const half_t* __restrict__ Bhi, const half_t* __restrict__ Blo,
    int npad, int bstride, int numB, int rowsPerBatch,
    float* __restrict__ C, int ldc,
    int M, int Nvalid)
{
    const int waveId  = threadIdx.x >> 5;
    const int lane    = threadIdx.x & 31;
    const int tilesN  = npad >> 4;
    const int groupsN = tilesN / NT;
    const int tilesM  = M >> 4;
    const int group   = blockIdx.x * (blockDim.x >> 5) + waveId;
    if (group >= tilesM * groupsN) return;   // wave-uniform: EXEC stays all-ones
    const int tm  = group / groupsN;
    const int tn0 = (group % groupsN) * NT;
    const int bsel = ((tm * 16) / rowsPerBatch) % numB;
    const v16h* bhp = (const v16h*)(Bhi + (size_t)bsel * bstride);
    const v16h* blp = (const v16h*)(Blo + (size_t)bsel * bstride);

    const int arow = tm * 16 + (lane & 15);
    const int q    = (lane >> 4) * 4;          // float4 index base within 32-k group
    const float4* ap4 = (const float4*)(A + (size_t)arow * lda);

    v8f acc[NT];
#pragma unroll
    for (int t = 0; t < NT; ++t) acc[t] = (v8f){};

#pragma unroll
    for (int kv = 0; kv < KDIMC / 32; ++kv) {
        // ---- A fragment: 16 contiguous f32 (k = 16h + s), split once ----
        float4 a0 = ap4[kv * 8 + q];
        float4 a1 = ap4[kv * 8 + q + 1];
        float4 a2 = ap4[kv * 8 + q + 2];
        float4 a3 = ap4[kv * 8 + q + 3];
        float af[16] = { a0.x, a0.y, a0.z, a0.w, a1.x, a1.y, a1.z, a1.w,
                         a2.x, a2.y, a2.z, a2.w, a3.x, a3.y, a3.z, a3.w };
        v16h ahi, alo;
#pragma unroll
        for (int e = 0; e < 16; ++e) {
            half_t h = (half_t)af[e];
            ahi[e] = h;
            alo[e] = (half_t)(af[e] - (float)h);
        }
        // ---- prefetch all NT B-fragment pairs (one 32-byte v16h per lane each) ----
        v16h bhi[NT], blo[NT];
#pragma unroll
        for (int t = 0; t < NT; ++t) {
            int fidx = (kv * tilesN + tn0 + t) * 32 + lane;
            bhi[t] = bhp[fidx];
            blo[t] = blp[fidx];
        }
        // ---- three t-sweeps: consecutive WMMAs use different accumulators ----
#pragma unroll
        for (int t = 0; t < NT; ++t)
            acc[t] = __builtin_amdgcn_wmma_f32_16x16x32_f16(false, ahi, false, bhi[t], (short)0, acc[t], false, false);
#pragma unroll
        for (int t = 0; t < NT; ++t)
            acc[t] = __builtin_amdgcn_wmma_f32_16x16x32_f16(false, alo, false, bhi[t], (short)0, acc[t], false, false);
#pragma unroll
        for (int t = 0; t < NT; ++t)
            acc[t] = __builtin_amdgcn_wmma_f32_16x16x32_f16(false, ahi, false, blo[t], (short)0, acc[t], false, false);
    }
    const int rbase = tm * 16 + ((lane >> 4) << 3);
#pragma unroll
    for (int t = 0; t < NT; ++t) {
        int n = (tn0 + t) * 16 + (lane & 15);
        if (n < Nvalid) {
            float* cp = C + (size_t)rbase * ldc + n;
#pragma unroll
            for (int v = 0; v < 8; ++v) cp[(size_t)v * ldc] = acc[t][v];
        }
    }
}

// ---------------------------------------------------------------------------
// Depthwise 3x3 conv (SAME, zero pad) + bias + SiLU.
// Input:  xa = first 96 cols of xz (B*L, 192).   Output: xc (B, L, d)
// 96 threads over d -> fully coalesced reads/writes.
// ---------------------------------------------------------------------------
__global__ __launch_bounds__(96) void conv_silu(const float* __restrict__ xz,
                                                const float* __restrict__ cw,
                                                const float* __restrict__ cb,
                                                float* __restrict__ xc) {
    int b = blockIdx.x >> 12;
    int l = blockIdx.x & (LDIM - 1);
    int h = l >> 6, w = l & 63;
    int dd = threadIdx.x;
    float acc = cb[dd];
#pragma unroll
    for (int kh = -1; kh <= 1; ++kh) {
#pragma unroll
        for (int kw = -1; kw <= 1; ++kw) {
            int hh = h + kh, ww = w + kw;
            if (hh >= 0 && hh < HDIM && ww >= 0 && ww < WDIM) {
                float xv = xz[((size_t)b * LDIM + hh * WDIM + ww) * (2 * DINN) + dd];
                acc += cw[dd * 9 + (kh + 1) * 3 + (kw + 1)] * xv;
            }
        }
    }
    float s = acc / (1.f + __expf(-acc));   // SiLU
    xc[((size_t)b * LDIM + l) * DINN + dd] = s;
}

// ---------------------------------------------------------------------------
// Build 4 directional scan inputs, (b,k,l,d); coalesced gather from xc (B,L,d)
// ---------------------------------------------------------------------------
__device__ __forceinline__ int scan_src(int k, int l) {
    switch (k) {
        case 0: return l;
        case 1: return (l & 63) * 64 + (l >> 6);
        case 2: return LDIM - 1 - l;
        default: { int lr = LDIM - 1 - l; return (lr & 63) * 64 + (lr >> 6); }
    }
}

__global__ __launch_bounds__(96) void build_xs(const float* __restrict__ xc,
                                               float* __restrict__ xs) {
    int l = blockIdx.x & (LDIM - 1);
    int k = (blockIdx.x >> 12) & 3;
    int b = blockIdx.x >> 14;
    int dd = threadIdx.x;
    int src = scan_src(k, l);
    xs[(((size_t)(b * KDIR + k)) * LDIM + l) * DINN + dd] =
        xc[((size_t)b * LDIM + src) * DINN + dd];
}

// ---------------------------------------------------------------------------
// Selective scan, chunked (parallel-prefix over 64 chunks of 64 steps).
// ---------------------------------------------------------------------------
__device__ __forceinline__ float softplus_f(float x) {
    return fmaxf(x, 0.f) + log1pf(__expf(-fabsf(x)));
}

__global__ __launch_bounds__(96) void scan_pass1(
    const float* __restrict__ xdbl, const float* __restrict__ xs,
    const float* __restrict__ A_logs, const float* __restrict__ dt_w,
    const float* __restrict__ dt_b,
    float* __restrict__ Pbuf, float* __restrict__ Hbuf)
{
    __shared__ float sxd[CHLEN][CDBL];
    __shared__ float su[CHLEN][DINN];
    const int tid = threadIdx.x;
    const int chunk = blockIdx.x & 63;
    const int k = (blockIdx.x >> 6) & 3;
    const int b = blockIdx.x >> 8;
    const int bk = b * KDIR + k;
    const int l0 = chunk * CHLEN;
    const float* xdbase = xdbl + ((size_t)bk * LDIM + l0) * CDBL;
    const float* xsbase = xs   + ((size_t)bk * LDIM + l0) * DINN;
    for (int i = tid; i < CHLEN * CDBL; i += 96) sxd[i / CDBL][i % CDBL] = xdbase[i];
    for (int i = tid; i < CHLEN * DINN; i += 96) su[i / DINN][i % DINN] = xsbase[i];
    __syncthreads();

    const int dd = tid;
    const int kd = k * DINN + dd;
    float An[NST], dtw[RNK];
#pragma unroll
    for (int n = 0; n < NST; ++n) An[n] = -__expf(A_logs[kd * NST + n]);
#pragma unroll
    for (int r = 0; r < RNK; ++r) dtw[r] = dt_w[kd * RNK + r];
    const float dtb = dt_b[kd];

    float h[NST], P[NST];
#pragma unroll
    for (int n = 0; n < NST; ++n) { h[n] = 0.f; P[n] = 1.f; }

    for (int j = 0; j < CHLEN; ++j) {
        float dt = dtb;
#pragma unroll
        for (int r = 0; r < RNK; ++r) dt += dtw[r] * sxd[j][r];
        float delta = softplus_f(dt);
        float du = delta * su[j][dd];
#pragma unroll
        for (int n = 0; n < NST; ++n) {
            float dA = __expf(delta * An[n]);
            P[n] *= dA;
            h[n] = dA * h[n] + sxd[j][RNK + n] * du;
        }
    }
    size_t base = (((size_t)bk * NCHUNK + chunk) * DINN + dd) * NST;
#pragma unroll
    for (int n = 0; n < NST; ++n) { Pbuf[base + n] = P[n]; Hbuf[base + n] = h[n]; }
}

// Pass 2: sequential combine over the 64 chunks (in-place: Hbuf -> chunk-initial h)
__global__ void scan_pass2(const float* __restrict__ Pbuf, float* __restrict__ Hbuf) {
    int t = blockIdx.x * blockDim.x + threadIdx.x;   // B*K*d*N = 12288
    if (t >= BATCH * KDIR * DINN * NST) return;
    int n = t & 7;
    int dd = (t >> 3) % DINN;
    int bk = t / (DINN * NST);
    float hs = 0.f;
    for (int c = 0; c < NCHUNK; ++c) {
        size_t base = (((size_t)bk * NCHUNK + c) * DINN + dd) * NST + n;
        float p = Pbuf[base];
        float hl = Hbuf[base];
        Hbuf[base] = hs;          // becomes initial state for chunk c
        hs = p * hs + hl;
    }
}

// Pass 3: re-run chunks from correct initial state, emit y (in place over xs)
__global__ __launch_bounds__(96) void scan_pass3(
    const float* __restrict__ xdbl, float* __restrict__ xs /* in: u, out: y */,
    const float* __restrict__ A_logs, const float* __restrict__ dt_w,
    const float* __restrict__ dt_b, const float* __restrict__ Ds,
    const float* __restrict__ Hinit)
{
    __shared__ float sxd[CHLEN][CDBL];
    __shared__ float su[CHLEN][DINN];
    const int tid = threadIdx.x;
    const int chunk = blockIdx.x & 63;
    const int k = (blockIdx.x >> 6) & 3;
    const int b = blockIdx.x >> 8;
    const int bk = b * KDIR + k;
    const int l0 = chunk * CHLEN;
    const float* xdbase = xdbl + ((size_t)bk * LDIM + l0) * CDBL;
    float* xsbase = xs + ((size_t)bk * LDIM + l0) * DINN;
    for (int i = tid; i < CHLEN * CDBL; i += 96) sxd[i / CDBL][i % CDBL] = xdbase[i];
    for (int i = tid; i < CHLEN * DINN; i += 96) su[i / DINN][i % DINN] = xsbase[i];
    __syncthreads();

    const int dd = tid;
    const int kd = k * DINN + dd;
    float An[NST], dtw[RNK];
#pragma unroll
    for (int n = 0; n < NST; ++n) An[n] = -__expf(A_logs[kd * NST + n]);
#pragma unroll
    for (int r = 0; r < RNK; ++r) dtw[r] = dt_w[kd * RNK + r];
    const float dtb = dt_b[kd];
    const float Dk = Ds[kd];

    float h[NST];
    size_t base = (((size_t)bk * NCHUNK + chunk) * DINN + dd) * NST;
#pragma unroll
    for (int n = 0; n < NST; ++n) h[n] = Hinit[base + n];

    for (int j = 0; j < CHLEN; ++j) {
        float dt = dtb;
#pragma unroll
        for (int r = 0; r < RNK; ++r) dt += dtw[r] * sxd[j][r];
        float delta = softplus_f(dt);
        float u = su[j][dd];
        float du = delta * u;
        float y = Dk * u;
#pragma unroll
        for (int n = 0; n < NST; ++n) {
            float dA = __expf(delta * An[n]);
            h[n] = dA * h[n] + sxd[j][RNK + n] * du;
            y += h[n] * sxd[j][RNK + NST + n];
        }
        xsbase[(size_t)j * DINN + dd] = y;   // safe: u already staged in LDS
    }
}

// ---------------------------------------------------------------------------
// Combine the 4 directions (with index un-mapping), LayerNorm over d, SiLU gate
// ---------------------------------------------------------------------------
__global__ __launch_bounds__(96) void combine_ln_gate(
    const float* __restrict__ ys, const float* __restrict__ xz,
    const float* __restrict__ g, const float* __restrict__ be,
    float* __restrict__ out)
{
    __shared__ float red[DINN];
    __shared__ float stat[2];
    const int b  = blockIdx.x >> 12;
    const int lo = blockIdx.x & (LDIM - 1);
    const int dd = threadIdx.x;
    const int hh = lo >> 6, ww = lo & 63;
    const int l1 = ww * 64 + hh;               // wh-transpose inverse map
    float y = ys[(((size_t)(b * KDIR + 0)) * LDIM + lo) * DINN + dd]
            + ys[(((size_t)(b * KDIR + 1)) * LDIM + l1) * DINN + dd]
            + ys[(((size_t)(b * KDIR + 2)) * LDIM + (LDIM - 1 - lo)) * DINN + dd]
            + ys[(((size_t)(b * KDIR + 3)) * LDIM + (LDIM - 1 - l1)) * DINN + dd];
    red[dd] = y;
    __syncthreads();
    if (dd == 0) { float s = 0.f; for (int i = 0; i < DINN; ++i) s += red[i]; stat[0] = s / DINN; }
    __syncthreads();
    float m = stat[0];
    float cdev = y - m;
    red[dd] = cdev * cdev;
    __syncthreads();
    if (dd == 0) { float s = 0.f; for (int i = 0; i < DINN; ++i) s += red[i]; stat[1] = s / DINN; }
    __syncthreads();
    float yo = cdev * rsqrtf(stat[1] + 1e-5f) * g[dd] + be[dd];
    float z = xz[((size_t)b * LDIM + lo) * (2 * DINN) + DINN + dd];
    float sz = z / (1.f + __expf(-z));
    out[((size_t)b * LDIM + lo) * DINN + dd] = yo * sz;
}

// ---------------------------------------------------------------------------
// Host side
// ---------------------------------------------------------------------------
struct LayerParams {
    const float *in_w, *conv_w, *conv_b, *xproj_w, *dt_w, *dt_b, *A_logs, *Ds, *ln_g, *ln_b, *out_w;
};

template <int NT>
static void launch_gemm(const float* A, int lda, const half_t* Bhi, const half_t* Blo,
                        int npad, int bstride, int numB, int rowsPerBatch,
                        float* C, int ldc, int M, int Nvalid, hipStream_t s) {
    int groups = (M / 16) * (npad / 16 / NT);
    int blocks = (groups + 7) / 8;    // 8 waves (256 threads) per block
    gemm_splitf16<NT><<<blocks, 256, 0, s>>>(A, lda, Bhi, Blo, npad, bstride, numB,
                                             rowsPerBatch, C, ldc, M, Nvalid);
}

extern "C" void kernel_launch(void* const* d_in, const int* in_sizes, int n_in,
                              void* d_out, int out_size, void* d_ws, size_t ws_size,
                              hipStream_t stream) {
    (void)in_sizes; (void)n_in; (void)out_size; (void)ws_size;

    // Workspace carve-up (floats). Total ~69 MB -> L2 resident on MI455X.
    float* x_in = (float*)d_ws;                        // B*L*C      = 1,572,864
    float* xz   = x_in + (size_t)BATCH * LDIM * CDIM;  // B*L*2d     = 3,145,728
    float* xc   = xz   + (size_t)BATCH * LDIM * 2 * DINN;        // 1,572,864
    float* xs   = xc   + (size_t)BATCH * DINN * LDIM;  // B*K*L*d    = 6,291,456
    float* xdbl = xs   + (size_t)BATCH * KDIR * LDIM * DINN;     // 1,441,792
    float* Pb   = xdbl + (size_t)BATCH * KDIR * LDIM * CDBL;     //   786,432
    float* Hb   = Pb   + (size_t)BATCH * KDIR * NCHUNK * DINN * NST;
    float* yln  = Hb   + (size_t)BATCH * KDIR * NCHUNK * DINN * NST; // 1,572,864
    half_t* w1h = (half_t*)(yln + (size_t)BATCH * LDIM * DINN);
    half_t* w1l = w1h + DINN * 2 * DINN;               // 96*192
    half_t* w2h = w1l + DINN * 2 * DINN;
    half_t* w2l = w2h + KDIR * DINN * 32;              // 4*96*32
    half_t* w3h = w2l + KDIR * DINN * 32;
    half_t* w3l = w3h + DINN * CDIM;                   // 96*96

    const size_t TOT = (size_t)BATCH * LDIM * CDIM;    // 1,572,864

    // Input NCHW -> (B*L, C)
    nchw_to_nlc<<<(TOT + 255) / 256, 256, 0, stream>>>((const float*)d_in[0], x_in);

    for (int layer = 0; layer < 2; ++layer) {
        const int o = 1 + layer * 11;
        LayerParams p = {
            (const float*)d_in[o + 0], (const float*)d_in[o + 1], (const float*)d_in[o + 2],
            (const float*)d_in[o + 3], (const float*)d_in[o + 4], (const float*)d_in[o + 5],
            (const float*)d_in[o + 6], (const float*)d_in[o + 7], (const float*)d_in[o + 8],
            (const float*)d_in[o + 9], (const float*)d_in[o + 10]
        };

        // Split + fragment-pack weights for the WMMA path
        prep_transpose_split<<<(DINN * 2 * DINN + 255) / 256, 256, 0, stream>>>(
            p.in_w, w1h, w1l, 2 * DINN, CDIM, 2 * DINN);           // B1: 96 x 192
        prep_w2<<<(KDIR * DINN * 32 + 255) / 256, 256, 0, stream>>>(p.xproj_w, w2h, w2l);
        prep_transpose_split<<<(DINN * CDIM + 255) / 256, 256, 0, stream>>>(
            p.out_w, w3h, w3l, CDIM, DINN, CDIM);                  // B3: 96 x 96

        // 1) xz = x @ in_w^T   (16384 x 96 x 192), 4 tile-cols per wave
        launch_gemm<4>(x_in, CDIM, w1h, w1l, 2 * DINN, 0, 1, 1 << 30,
                       xz, 2 * DINN, BATCH * LDIM, 2 * DINN, stream);

        // 2) depthwise conv 3x3 + SiLU -> xc (B,L,d)
        conv_silu<<<BATCH * LDIM, 96, 0, stream>>>(xz, p.conv_w, p.conv_b, xc);

        // 3) 4-direction scan inputs, (b,k,l,d)
        build_xs<<<BATCH * KDIR * LDIM, 96, 0, stream>>>(xc, xs);

        // 4) x_dbl = xs @ xproj_w[k]^T   (65536 x 96 x 22, per-k weights)
        launch_gemm<2>(xs, DINN, w2h, w2l, 32, DINN * 32, KDIR, LDIM,
                       xdbl, CDBL, BATCH * KDIR * LDIM, CDBL, stream);

        // 5) chunked selective scan
        scan_pass1<<<BATCH * KDIR * NCHUNK, 96, 0, stream>>>(
            xdbl, xs, p.A_logs, p.dt_w, p.dt_b, Pb, Hb);
        scan_pass2<<<(BATCH * KDIR * DINN * NST + 255) / 256, 256, 0, stream>>>(Pb, Hb);
        scan_pass3<<<BATCH * KDIR * NCHUNK, 96, 0, stream>>>(
            xdbl, xs, p.A_logs, p.dt_w, p.dt_b, p.Ds, Hb);

        // 6) combine 4 directions + LayerNorm + SiLU(z) gate
        combine_ln_gate<<<BATCH * LDIM, 96, 0, stream>>>(xs, xz, p.ln_g, p.ln_b, yln);

        // 7) out = yln @ out_w^T  (16384 x 96 x 96) -> next layer input
        launch_gemm<3>(yln, DINN, w3h, w3l, CDIM, 0, 1, 1 << 30,
                       x_in, CDIM, BATCH * LDIM, CDIM, stream);
    }

    // (B*L, C) -> NCHW output
    nlc_to_nchw<<<(TOT + 255) / 256, 256, 0, stream>>>(x_in, (float*)d_out);
}
